// WindTransformer_71451075937017
// MI455X (gfx1250) — matrix operand order repaired and checked by reading, not compiled, not run on hardware
//
#include <hip/hip_runtime.h>

// Swin window attention, fused, f16 WMMA + f32 accumulate for gfx1250 (MI455X).
// B_=1024, N=288, C=96, H=4, d=24, nW=128.

typedef __attribute__((ext_vector_type(16))) _Float16 v16h;
typedef __attribute__((ext_vector_type(8)))  float    v8f;

#define QK_SCALE 0.20412414523193154f   // 24^-0.5

static __device__ __forceinline__ v8f zero8() {
  v8f z;
#pragma unroll
  for (int i = 0; i < 8; ++i) z[i] = 0.0f;
  return z;
}

static __device__ __forceinline__ v8f wmma_f16(v16h a, v16h b, v8f c) {
  // v_wmma_f32_16x16x32_f16
  return __builtin_amdgcn_wmma_f32_16x16x32_f16(false, a, false, b, (short)0, c,
                                                false, false);
}

// A-matrix 16x32 f16 fragment from row-major LDS tile (ld halves per row).
// Layout: lane m=L&15, hi=L>>4; half i -> K = 16*hi + i (16 contiguous halves).
static __device__ __forceinline__ v16h load_frag_a(const _Float16* __restrict__ base,
                                                   int ld, int lane) {
  const int m  = lane & 15;
  const int hi = (lane >> 4) & 1;
  const _Float16* row = base + m * ld;
  v16h a;
#pragma unroll
  for (int i = 0; i < 16; ++i) {
    const int vg = i >> 1, sub = i & 1;
    const int kk = 16 * hi + 2 * vg + sub;
    a[i] = row[kk];
  }
  return a;
}

// B-matrix 32x16 f16 fragment with B[k][n] = src[n][k] (src row-major over n, ld halves).
// Layout: lane n=L&15, hi=L>>4; half i -> K = 16*hi + i  (16 contiguous halves).
static __device__ __forceinline__ v16h load_frag_b_nk(const _Float16* __restrict__ base,
                                                      int ld, int lane) {
  const int n  = lane & 15;
  const int hi = (lane >> 4) & 1;
  const _Float16* row = base + n * ld + 16 * hi;
  v16h b;
#pragma unroll
  for (int i = 0; i < 16; ++i) b[i] = row[i];
  return b;
}

// ---------------------------------------------------------------------------
// Kernel 1: qkv = x @ qkv_w.T + qkv_b  -> f16 q(scaled)/k/v  [b][h][n][24]
// grid (2304, 3), block 256 (8 waves). blockIdx.y selects q/k/v column group.
// ---------------------------------------------------------------------------
__global__ __launch_bounds__(256)
void qkv_gemm_kernel(const float* __restrict__ x, const float* __restrict__ qkv_w,
                     const float* __restrict__ qkv_b,
                     _Float16* __restrict__ q, _Float16* __restrict__ k,
                     _Float16* __restrict__ v) {
  __shared__ _Float16 Xs[128 * 96];
  __shared__ _Float16 Ws[96 * 96];
  const int tid  = threadIdx.x;
  const int lane = tid & 31;
  const int wv   = tid >> 5;            // 0..7
  const int row0 = blockIdx.x * 128;    // global row in (B_*N, .)
  const int which = blockIdx.y;         // 0=q 1=k 2=v
  const int cbase = which * 96;

  for (int i = tid; i < 128 * 96; i += 256)
    Xs[i] = (_Float16)x[(size_t)row0 * 96 + i];
  for (int i = tid; i < 96 * 96; i += 256)
    Ws[i] = (_Float16)qkv_w[(size_t)cbase * 96 + i];
  __syncthreads();

  v8f acc[6];
#pragma unroll
  for (int ct = 0; ct < 6; ++ct) acc[ct] = zero8();

#pragma unroll
  for (int kt = 0; kt < 3; ++kt) {
    const v16h A = load_frag_a(&Xs[(wv * 16) * 96 + kt * 32], 96, lane);
#pragma unroll
    for (int ct = 0; ct < 6; ++ct) {
      const v16h B = load_frag_b_nk(&Ws[(ct * 16) * 96 + kt * 32], 96, lane);
      acc[ct] = wmma_f16(A, B, acc[ct]);
    }
  }

  const int nloc = lane & 15;
  const int hi   = (lane >> 4) & 1;
  _Float16* dst  = (which == 0) ? q : (which == 1) ? k : v;
  const float sc = (which == 0) ? QK_SCALE : 1.0f;
#pragma unroll
  for (int ct = 0; ct < 6; ++ct) {
    const int lc = ct * 16 + nloc;      // 0..95 within the 96-col group
    const int h  = lc / 24;
    const int dd = lc % 24;
    const float bias = qkv_b[cbase + lc];
#pragma unroll
    for (int r = 0; r < 8; ++r) {
      const int row = row0 + wv * 16 + r + 8 * hi;
      const int b   = row / 288;
      const int n   = row % 288;
      dst[(((size_t)b * 4 + h) * 288 + n) * 24 + dd] =
          (_Float16)((acc[ct][r] + bias) * sc);
    }
  }
}

// ---------------------------------------------------------------------------
// Kernel 2: fused flash attention per (b,h). grid 4096, block 192 (6 waves).
// Each wave handles 3 strips of 16 query rows; online softmax over 9 x 32 keys.
// ---------------------------------------------------------------------------
__global__ __launch_bounds__(192)
void attn_kernel(const _Float16* __restrict__ Q, const _Float16* __restrict__ K,
                 const _Float16* __restrict__ V, const float* __restrict__ mask,
                 _Float16* __restrict__ O) {
  __shared__ _Float16 Qs[288 * 32];     // [token][dk pad32]
  __shared__ _Float16 Ks[288 * 32];     // [token][dk pad32]
  __shared__ _Float16 Vs[32 * 288];     // transposed: [dd pad32][token]
  __shared__ _Float16 Ps[6 * 16 * 32];  // per-wave P tile (D->A layout bounce)

  const int tid  = threadIdx.x;
  const int lane = tid & 31;
  const int wv   = tid >> 5;            // 0..5
  const int bh   = blockIdx.x;          // b*4 + h
  const int b    = bh >> 2;
  const int h    = bh & 3;
  const size_t base = (size_t)bh * 288 * 24;

  const _Float16* Qg = Q + base;
  const _Float16* Kg = K + base;
  const _Float16* Vg = V + base;
  for (int i = tid; i < 288 * 24; i += 192) {   // 36 iters, uniform
    const int n = i / 24, dk = i % 24;
    Qs[n * 32 + dk] = Qg[i];
    Ks[n * 32 + dk] = Kg[i];
    Vs[dk * 288 + n] = Vg[i];
  }
  for (int i = tid; i < 288 * 8; i += 192) {    // zero K-dim pad
    const int n = i >> 3, p = i & 7;
    Qs[n * 32 + 24 + p] = (_Float16)0.0f;
    Ks[n * 32 + 24 + p] = (_Float16)0.0f;
  }
  for (int i = tid; i < 8 * 288; i += 192)      // zero V dd pad rows
    Vs[24 * 288 + i] = (_Float16)0.0f;
  __syncthreads();

  const int nloc = lane & 15;
  const int hi   = (lane >> 4) & 1;
  const float* mb0 = mask + (size_t)(b & 127) * 288 * 288;
  _Float16* Pw = &Ps[wv * 16 * 32];

  for (int s = 0; s < 3; ++s) {
    const int strip = wv * 3 + s;       // 0..17
    const int row0  = strip * 16;
    const v16h Aq = load_frag_a(&Qs[row0 * 32], 32, lane);
    v8f o0 = zero8(), o1 = zero8();
    float mrow[8], lrow[8];
#pragma unroll
    for (int r = 0; r < 8; ++r) { mrow[r] = -1.0e30f; lrow[r] = 0.0f; }
    const float* mrb = mb0 + (size_t)row0 * 288;

    for (int jt = 0; jt < 9; ++jt) {
      const int j0 = jt * 32;
      const v16h Bk0 = load_frag_b_nk(&Ks[j0 * 32], 32, lane);
      const v16h Bk1 = load_frag_b_nk(&Ks[(j0 + 16) * 32], 32, lane);
      v8f s0 = wmma_f16(Aq, Bk0, zero8());
      v8f s1 = wmma_f16(Aq, Bk1, zero8());

      float p0[8], p1[8];
#pragma unroll
      for (int r = 0; r < 8; ++r) {
        const float* mp = mrb + (size_t)(r + 8 * hi) * 288 + j0 + nloc;
        const float a0 = s0[r] + mp[0];
        const float a1 = s1[r] + mp[16];
        float t = fmaxf(a0, a1);
        t = fmaxf(t, __shfl_xor(t, 1, 32));   // reduce across the 16 n-lanes
        t = fmaxf(t, __shfl_xor(t, 2, 32));
        t = fmaxf(t, __shfl_xor(t, 4, 32));
        t = fmaxf(t, __shfl_xor(t, 8, 32));
        const float nm = fmaxf(mrow[r], t);
        const float sc = __expf(mrow[r] - nm);
        mrow[r] = nm;
        p0[r] = __expf(a0 - nm);
        p1[r] = __expf(a1 - nm);
        float ps = p0[r] + p1[r];
        ps += __shfl_xor(ps, 1, 32);
        ps += __shfl_xor(ps, 2, 32);
        ps += __shfl_xor(ps, 4, 32);
        ps += __shfl_xor(ps, 8, 32);
        lrow[r] = lrow[r] * sc + ps;
        o0[r] *= sc;
        o1[r] *= sc;
      }
      // D-layout (n=lane&15, m=r+8*hi) -> LDS -> A-layout for P@V
#pragma unroll
      for (int r = 0; r < 8; ++r) {
        Pw[(r + 8 * hi) * 32 + nloc]      = (_Float16)p0[r];
        Pw[(r + 8 * hi) * 32 + 16 + nloc] = (_Float16)p1[r];
      }
      asm volatile("s_wait_dscnt 0x0" ::: "memory");  // wave-local LDS RAW fence
      const v16h Ap  = load_frag_a(Pw, 32, lane);
      const v16h Bv0 = load_frag_b_nk(&Vs[0 * 288 + j0], 288, lane);
      const v16h Bv1 = load_frag_b_nk(&Vs[16 * 288 + j0], 288, lane);
      o0 = wmma_f16(Ap, Bv0, o0);
      o1 = wmma_f16(Ap, Bv1, o1);
    }

    // epilogue: normalize and scatter to (B_, N, H*d) f16 staging
#pragma unroll
    for (int r = 0; r < 8; ++r) {
      const float inv = 1.0f / lrow[r];
      const int n = row0 + r + 8 * hi;
      _Float16* op = O + ((size_t)b * 288 + n) * 96 + h * 24;
      op[nloc] = (_Float16)(o0[r] * inv);
      if (nloc < 8) op[16 + nloc] = (_Float16)(o1[r] * inv);  // dd 16..23 only
    }
  }
}

// ---------------------------------------------------------------------------
// Kernel 3: out = attn @ proj_w.T + proj_b (f32 out). grid 2304, block 256.
// ---------------------------------------------------------------------------
__global__ __launch_bounds__(256)
void proj_gemm_kernel(const _Float16* __restrict__ a, const float* __restrict__ proj_w,
                      const float* __restrict__ proj_b, float* __restrict__ out) {
  __shared__ _Float16 As[128 * 96];
  __shared__ _Float16 Ws[96 * 96];
  const int tid  = threadIdx.x;
  const int lane = tid & 31;
  const int wv   = tid >> 5;
  const int row0 = blockIdx.x * 128;

  for (int i = tid; i < 128 * 96; i += 256)
    As[i] = a[(size_t)row0 * 96 + i];
  for (int i = tid; i < 96 * 96; i += 256)
    Ws[i] = (_Float16)proj_w[i];          // proj_w is [c][k] row-major already
  __syncthreads();

  v8f acc[6];
#pragma unroll
  for (int ct = 0; ct < 6; ++ct) acc[ct] = zero8();

#pragma unroll
  for (int kt = 0; kt < 3; ++kt) {
    const v16h A = load_frag_a(&As[(wv * 16) * 96 + kt * 32], 96, lane);
#pragma unroll
    for (int ct = 0; ct < 6; ++ct) {
      const v16h B = load_frag_b_nk(&Ws[(ct * 16) * 96 + kt * 32], 96, lane);
      acc[ct] = wmma_f16(A, B, acc[ct]);
    }
  }

  const int nloc = lane & 15;
  const int hi   = (lane >> 4) & 1;
#pragma unroll
  for (int ct = 0; ct < 6; ++ct) {
    const int c = ct * 16 + nloc;
    const float bias = proj_b[c];
#pragma unroll
    for (int r = 0; r < 8; ++r)
      out[(size_t)(row0 + wv * 16 + r + 8 * hi) * 96 + c] = acc[ct][r] + bias;
  }
}

// ---------------------------------------------------------------------------
extern "C" void kernel_launch(void* const* d_in, const int* in_sizes, int n_in,
                              void* d_out, int out_size, void* d_ws, size_t ws_size,
                              hipStream_t stream) {
  (void)in_sizes; (void)n_in; (void)out_size; (void)ws_size;
  const float* x      = (const float*)d_in[0];
  const float* mask   = (const float*)d_in[1];
  const float* qkv_w  = (const float*)d_in[2];
  const float* qkv_b  = (const float*)d_in[3];
  const float* proj_w = (const float*)d_in[4];
  const float* proj_b = (const float*)d_in[5];
  float* out = (float*)d_out;

  // f16 workspace: Q | K | V | attn_out, each 1024*4*288*24 halves (56.6 MB)
  const size_t qkv_elems = (size_t)1024 * 4 * 288 * 24;
  _Float16* Q  = (_Float16*)d_ws;
  _Float16* Kp = Q + qkv_elems;
  _Float16* Vp = Kp + qkv_elems;
  _Float16* AO = Vp + qkv_elems;

  qkv_gemm_kernel<<<dim3(2304, 3), 256, 0, stream>>>(x, qkv_w, qkv_b, Q, Kp, Vp);
  attn_kernel<<<dim3(4096), 192, 0, stream>>>(Q, Kp, Vp, mask, AO);
  proj_gemm_kernel<<<dim3(2304), 256, 0, stream>>>(AO, proj_w, proj_b, out);
}
